// AqtEinsum_19481971655319
// MI455X (gfx1250) — 8-bit matrix operand order repaired and checked by reading, compiled, not run on hardware
//
#include <hip/hip_runtime.h>

typedef __attribute__((ext_vector_type(8))) int          v8i;
typedef __attribute__((ext_vector_type(4))) int          i4;
typedef __attribute__((ext_vector_type(4))) float        f4;
typedef __attribute__((ext_vector_type(4))) unsigned int u32x4;
typedef __attribute__((ext_vector_type(8))) int          i32x8;
typedef __attribute__((ext_vector_type(4))) int          i32x4;

#define B_DIM 4
#define S_DIM 2048
#define M_TOT 8192
#define K_TOT 4096
#define N_TOT 16384

#define BM 128
#define BN 256
#define BK 64
#define NSTEP (K_TOT / BK)
#define LDA 80
#define LDB 272

#ifndef __has_builtin
#define __has_builtin(x) 0
#endif
#if __has_builtin(__builtin_amdgcn_tensor_load_to_lds) && \
    __has_builtin(__builtin_amdgcn_s_wait_tensorcnt)
#define USE_TDM 1
#else
#define USE_TDM 0
#endif

__global__ __launch_bounds__(256) void q_lhs_kernel(const float* __restrict__ lhs,
                                                    signed char* __restrict__ q,
                                                    float* __restrict__ scale) {
  const int row = blockIdx.x;
  const int t   = threadIdx.x;
  const float* src = lhs + (size_t)row * K_TOT;

  f4 x[4];
  float amax = 0.0f;
#pragma unroll
  for (int i = 0; i < 4; ++i) {
    x[i] = ((const f4*)src)[i * 256 + t];
#pragma unroll
    for (int j = 0; j < 4; ++j) amax = fmaxf(amax, fabsf(x[i][j]));
  }

  __shared__ float red[256];
  red[t] = amax;
  __syncthreads();
#pragma unroll
  for (int s = 128; s > 0; s >>= 1) {
    if (t < s) red[t] = fmaxf(red[t], red[t + s]);
    __syncthreads();
  }
  const float am  = red[0];
  const float sc  = (am == 0.0f) ? 1.0f : am * (1.0f / 127.0f);
  const float inv = 1.0f / sc;
  if (t == 0) scale[row] = sc;

  signed char* dst = q + (size_t)row * K_TOT;
#pragma unroll
  for (int i = 0; i < 4; ++i) {
    int packed = 0;
#pragma unroll
    for (int j = 0; j < 4; ++j) {
      float r = rintf(x[i][j] * inv);
      r = fminf(127.0f, fmaxf(-127.0f, r));
      packed |= ((int)r & 0xFF) << (8 * j);
    }
    ((int*)dst)[i * 256 + t] = packed;
  }
}

__global__ __launch_bounds__(256) void q_rhs_kernel(const float* __restrict__ rhs,
                                                    signed char* __restrict__ q,
                                                    float* __restrict__ scale) {
  const int f = blockIdx.x * 256 + threadIdx.x;
  float amax = 0.0f;
  for (int d = 0; d < K_TOT; d += 4) {
#pragma unroll
    for (int u = 0; u < 4; ++u)
      amax = fmaxf(amax, fabsf(rhs[(size_t)(d + u) * N_TOT + f]));
  }
  const float sc  = (amax == 0.0f) ? 1.0f : amax * (1.0f / 127.0f);
  const float inv = 1.0f / sc;
  scale[f] = sc;
  for (int d = 0; d < K_TOT; d += 4) {
#pragma unroll
    for (int u = 0; u < 4; ++u) {
      float r = rintf(rhs[(size_t)(d + u) * N_TOT + f] * inv);
      r = fminf(127.0f, fmaxf(-127.0f, r));
      q[(size_t)(d + u) * N_TOT + f] = (signed char)(int)r;
    }
  }
}

#if USE_TDM
__device__ __forceinline__ void tdm_load_2d(unsigned lds_addr, const void* gptr,
                                            unsigned tile_w, unsigned tile_h,
                                            unsigned long long stride_elems,
                                            unsigned pad_interval, unsigned pad_amount) {
  const unsigned long long ga = (unsigned long long)(size_t)gptr;
  u32x4 g0;
  g0.x = 1u;
  g0.y = lds_addr;
  g0.z = (unsigned)ga;
  g0.w = (unsigned)((ga >> 32) & 0x01FFFFFFull)
         | (2u << 30);
  i32x8 g1;
  g1[0] = (int)((1u << 20)
                | (pad_interval << 22) | (pad_amount << 25));
  g1[1] = (int)((tile_w & 0xFFFFu) << 16);
  g1[2] = (int)((tile_w >> 16) | ((tile_h & 0xFFFFu) << 16));
  g1[3] = (int)((tile_h >> 16) | (tile_w << 16));
  g1[4] = (int)tile_h;
  g1[5] = (int)(stride_elems & 0xFFFFFFFFull);
  g1[6] = (int)((stride_elems >> 32) & 0xFFFFull);
  g1[7] = 0;
  const i32x4 z4 = {0, 0, 0, 0};
  const i32x8 z8 = {0, 0, 0, 0, 0, 0, 0, 0};
  __builtin_amdgcn_tensor_load_to_lds(g0, g1, z4, z4, z8, 0);
}
#endif

__global__ __launch_bounds__(256) void gemm_iu8_kernel(const signed char* __restrict__ A,
                                                       const signed char* __restrict__ B,
                                                       const float* __restrict__ sA,
                                                       const float* __restrict__ sB,
                                                       float* __restrict__ out) {
  __shared__ __align__(16) signed char lA[2][BM * LDA];
  __shared__ __align__(16) signed char lB[2][BK * LDB];

  const int t     = threadIdx.x;
  const int lane  = t & 31;
  const int wid   = t >> 5;
  const int wm    = wid >> 2;
  const int wn    = wid & 3;
  const int tileM = blockIdx.y * BM;
  const int tileN = blockIdx.x * BN;

  v8i acc[4][4] = {};

#if !USE_TDM
  const int a_r = t >> 2;
  const int a_c = (t & 3) * 16;
  const int b_r = t >> 4;
  const int b_c = (t & 15) * 16;
#endif

  auto fill = [&](int buf, int k0) {
#if USE_TDM
    if (wid == 0) {
      tdm_load_2d((unsigned)(size_t)(void*)&lA[buf][0],
                  A + (size_t)tileM * K_TOT + k0,
                   BK,  BM,  K_TOT,
                   3,  3);
      tdm_load_2d((unsigned)(size_t)(void*)&lB[buf][0],
                  B + (size_t)k0 * N_TOT + tileN,
                   BN,  BK,  N_TOT,
                   5,  3);
    }
#else
#pragma unroll
    for (int p = 0; p < 2; ++p) {
      const int ar = a_r + p * 64;
      *(i4*)(&lA[buf][0] + ar * LDA + a_c) =
          *(const i4*)(A + (size_t)(tileM + ar) * K_TOT + k0 + a_c);
    }
#pragma unroll
    for (int p = 0; p < 4; ++p) {
      const int br = b_r + p * 16;
      *(i4*)(&lB[buf][0] + br * LDB + b_c) =
          *(const i4*)(B + (size_t)(k0 + br) * N_TOT + tileN + b_c);
    }
    if (k0 + BK < K_TOT) {
      __builtin_prefetch(A + (size_t)(tileM + a_r) * K_TOT + k0 + BK + a_c, 0, 1);
      __builtin_prefetch(B + (size_t)(k0 + BK + b_r) * N_TOT + tileN + b_c, 0, 1);
    }
#endif
  };

  fill(0, 0);

  for (int s = 0; s < NSTEP; ++s) {
    const int buf = s & 1;
    if (s + 1 < NSTEP) fill(buf ^ 1, (s + 1) * BK);

#if USE_TDM
    if (wid == 0) {
      if (s + 1 < NSTEP) __builtin_amdgcn_s_wait_tensorcnt(2);
      else               __builtin_amdgcn_s_wait_tensorcnt(0);
    }
#endif
    __syncthreads();

    const signed char* cA = &lA[buf][0];
    const signed char* cB = &lB[buf][0];

    const int ks    = (lane >> 4) * 8;
    const int mrow0 = wm * 64 + (lane & 15);
    v8i afrag[4];
#pragma unroll
    for (int mt = 0; mt < 4; ++mt) {
      const signed char* ap = cA + (mrow0 + mt * 16) * LDA;
      union { long long d[4]; v8i v; } u;
      u.d[0] = *(const long long*)(ap + 0  + ks);
      u.d[1] = *(const long long*)(ap + 16 + ks);
      u.d[2] = *(const long long*)(ap + 32 + ks);
      u.d[3] = *(const long long*)(ap + 48 + ks);
      afrag[mt] = u.v;
    }

    v8i bfrag[4];
#pragma unroll
    for (int nt = 0; nt < 4; ++nt) {
      const int ncol = wn * 64 + nt * 16;
      union { signed char b[32]; v8i v; } u;
      const signed char* bp = cB + ncol + (lane & 15);
#pragma unroll
      for (int s2 = 0; s2 < 32; ++s2)
        u.b[s2] = bp[(16 * (s2 >> 3) + ks + (s2 & 7)) * LDB];
      bfrag[nt] = u.v;
    }

#pragma unroll
    for (int mt = 0; mt < 4; ++mt)
#pragma unroll
      for (int nt = 0; nt < 4; ++nt)
        acc[mt][nt] = __builtin_amdgcn_wmma_i32_16x16x64_iu8(
             true, afrag[mt],  true, bfrag[nt],
            acc[mt][nt],  false,  false);

    __syncthreads();
  }

  const int nlo = lane & 15;
  const int mhi = (lane >> 4) * 8;
#pragma unroll
  for (int nt = 0; nt < 4; ++nt) {
    const int col  = tileN + wn * 64 + nt * 16 + nlo;
    const float sb = sB[col];
#pragma unroll
    for (int mt = 0; mt < 4; ++mt) {
      const int rbase = tileM + wm * 64 + mt * 16 + mhi;
#pragma unroll
      for (int r = 0; r < 8; ++r) {
        const int row = rbase + r;
        out[(size_t)row * N_TOT + col] = (float)acc[mt][nt][r] * sA[row] * sb;
      }
    }
  }
}

extern "C" void kernel_launch(void* const* d_in, const int* in_sizes, int n_in,
                              void* d_out, int out_size, void* d_ws, size_t ws_size,
                              hipStream_t stream) {
  (void)in_sizes; (void)n_in; (void)out_size; (void)ws_size;
  const float* lhs = (const float*)d_in[0];
  const float* rhs = (const float*)d_in[1];
  float* out = (float*)d_out;

  char* ws = (char*)d_ws;
  signed char* qL = (signed char*)ws;
  signed char* qR = qL + (size_t)M_TOT * K_TOT;
  float* sL = (float*)(qR + (size_t)K_TOT * N_TOT);
  float* sR = sL + M_TOT;

  q_lhs_kernel<<<M_TOT, 256, 0, stream>>>(lhs, qL, sL);
  q_rhs_kernel<<<N_TOT / 256, 256, 0, stream>>>(rhs, qR, sR);

  dim3 grid(N_TOT / BN, M_TOT / BM);
  gemm_iu8_kernel<<<grid, 256, 0, stream>>>(qL, qR, sL, sR, out);
}
